// GestureTransformer_36344013259304
// MI455X (gfx1250) — matrix operand order repaired and checked by reading, not compiled, not run on hardware
//
#include <hip/hip_runtime.h>
#include <hip/hip_bf16.h>
#include <math.h>

// ---------------------------------------------------------------------------
// Types for CDNA5 WMMA (wave32, 16x16x32 bf16 -> f32)
// ---------------------------------------------------------------------------
typedef __bf16 bf16;
typedef __attribute__((ext_vector_type(16))) __bf16 v16bf;
typedef __attribute__((ext_vector_type(8)))  float  v8f;

union FragA { v16bf v; uint4 q[2]; };

__device__ __forceinline__ v8f wmma_bf16(v16bf a, v16bf b, v8f c) {
  // 8 args: (neg_a, A, neg_b, B, c_mod, C, reuse_a, reuse_b)
  return __builtin_amdgcn_wmma_f32_16x16x32_bf16(false, a, false, b, (short)0, c,
                                                 false, false);
}

__device__ __forceinline__ float silu_f(float x) { return x / (1.f + __expf(-x)); }

// ---------------------------------------------------------------------------
// Elementwise / normalization kernels (f32 math, bf16 outputs feeding WMMA)
// ---------------------------------------------------------------------------
__global__ __launch_bounds__(256) void cvt_k(const float* __restrict__ in,
                                             bf16* __restrict__ out, int n) {
  int i = blockIdx.x * 256 + threadIdx.x;
  if (i < n) out[i] = (bf16)in[i];
}

__global__ __launch_bounds__(256) void silu_cvt_k(const float* __restrict__ in,
                                                  bf16* __restrict__ out, int n) {
  int i = blockIdx.x * 256 + threadIdx.x;
  if (i < n) out[i] = (bf16)silu_f(in[i]);
}

// W[K][N] f32 -> Wt[N][K] bf16  (so WMMA B-fragment halves are contiguous 16B loads along K)
__global__ __launch_bounds__(256) void wtrans_k(const float* __restrict__ W,
                                                bf16* __restrict__ Wt, int K, int N) {
  int i = blockIdx.x * 256 + threadIdx.x;
  if (i < K * N) {
    int k = i / N, n = i - k * N;
    Wt[(size_t)n * K + k] = (bf16)W[i];
  }
}

// LayerNorm over last dim (width 512 or 256), one wave per row, bf16 out.
__global__ __launch_bounds__(256) void ln_k(const float* __restrict__ X,
                                            const float* __restrict__ g,
                                            const float* __restrict__ bta,
                                            bf16* __restrict__ out,
                                            int rows, int width) {
  int lane = threadIdx.x & 31;
  int row  = blockIdx.x * 8 + (threadIdx.x >> 5);
  if (row >= rows) return;
  int cnt = width >> 5;  // 8 or 16
  const float* xr = X + (size_t)row * width;
  float v[16];
  float s = 0.f;
  for (int e = 0; e < cnt; ++e) { v[e] = xr[e * 32 + lane]; s += v[e]; }
  for (int m = 16; m; m >>= 1) s += __shfl_xor(s, m, 32);
  float mean = s / (float)width;
  float vs = 0.f;
  for (int e = 0; e < cnt; ++e) { float d = v[e] - mean; vs += d * d; }
  for (int m = 16; m; m >>= 1) vs += __shfl_xor(vs, m, 32);
  float rs = rsqrtf(vs / (float)width + 1e-5f);
  for (int e = 0; e < cnt; ++e) {
    int idx = e * 32 + lane;
    out[(size_t)row * width + idx] = (bf16)((v[e] - mean) * rs * g[idx] + bta[idx]);
  }
}

// adaLN middle: h2 = silu( LN(y)*(1+scale_b) + shift_b ), width fixed 512.
__global__ __launch_bounds__(256) void modulate_k(const float* __restrict__ Y,
                                                  const float* __restrict__ E,
                                                  const float* __restrict__ g,
                                                  const float* __restrict__ bta,
                                                  bf16* __restrict__ out,
                                                  int rows, int tokPerB) {
  int lane = threadIdx.x & 31;
  int row  = blockIdx.x * 8 + (threadIdx.x >> 5);
  if (row >= rows) return;
  int bidx = row / tokPerB;
  const float* yr = Y + (size_t)row * 512;
  float v[16];
  float s = 0.f;
  for (int e = 0; e < 16; ++e) { v[e] = yr[e * 32 + lane]; s += v[e]; }
  for (int m = 16; m; m >>= 1) s += __shfl_xor(s, m, 32);
  float mean = s * (1.f / 512.f);
  float vs = 0.f;
  for (int e = 0; e < 16; ++e) { float d = v[e] - mean; vs += d * d; }
  for (int m = 16; m; m >>= 1) vs += __shfl_xor(vs, m, 32);
  float rs = rsqrtf(vs * (1.f / 512.f) + 1e-5f);
  for (int e = 0; e < 16; ++e) {
    int idx = e * 32 + lane;
    float h  = (v[e] - mean) * rs * g[idx] + bta[idx];
    float sc = E[(size_t)bidx * 1024 + idx];
    float sh = E[(size_t)bidx * 1024 + 512 + idx];
    h = h * (1.f + sc) + sh;
    out[(size_t)row * 512 + idx] = (bf16)silu_f(h);
  }
}

// ---------------------------------------------------------------------------
// WMMA GEMM: C[M,N] = A[M,K](bf16) @ Wt[N,K]^T(bf16) + bias, templated epilogue.
// Block: 256 thr = 8 waves as 2(M)x4(N); wave tile 32x64 -> block tile 64x256.
// Requires M%64==0, N%256==0, K%64==0 (true for every GEMM in this model).
// EPI: 0=plain, 1=exact GELU, 2=residual add (res==C allowed), 3=*0.125 (Q).
// k-loop is software-pipelined two deep: loads for step n+1 overlap the 8
// WMMAs of step n, so s_wait_loadcnt covers already-hidden latency.
// ---------------------------------------------------------------------------
template <int EPI, typename OT>
__global__ __launch_bounds__(256) void gemm_k(const bf16* __restrict__ A,
                                              const bf16* __restrict__ Wt,
                                              const float* __restrict__ bias,
                                              OT* C, const float* res,
                                              int M, int K, int Nn) {
  const int lane = threadIdx.x & 31;
  const int wv   = threadIdx.x >> 5;
  const int wm   = wv >> 2, wn = wv & 3;
  const int m16  = lane & 15, g = lane >> 4;
  const size_t mbase = (size_t)blockIdx.y * 64 + wm * 32;
  const size_t nbase = (size_t)blockIdx.x * 256 + wn * 64;

  const bf16* ar[2];
  const bf16* br[4];
#pragma unroll
  for (int i = 0; i < 2; ++i) ar[i] = A + (mbase + i * 16 + m16) * (size_t)K;
#pragma unroll
  for (int j = 0; j < 4; ++j) br[j] = Wt + (nbase + j * 16 + m16) * (size_t)K;

  v8f zero = {0.f, 0.f, 0.f, 0.f, 0.f, 0.f, 0.f, 0.f};
  v8f acc[2][4];
#pragma unroll
  for (int i = 0; i < 2; ++i)
#pragma unroll
    for (int j = 0; j < 4; ++j) acc[i][j] = zero;

  auto loadAB = [&](FragA (&a)[2], FragA (&b)[4], int kb) {
#pragma unroll
    for (int i = 0; i < 2; ++i) {
      const bf16* p = ar[i] + kb;
      a[i].q[0] = *(const uint4*)(p + 8 * g);        // elems 0..7  : K = 8g..
      a[i].q[1] = *(const uint4*)(p + 16 + 8 * g);   // elems 8..15 : K = 16+8g..
    }
#pragma unroll
    for (int j = 0; j < 4; ++j)
    { b[j].q[0] = *(const uint4*)(br[j] + kb + 8 * g);        // elems 0..7  : K = 8g..
      b[j].q[1] = *(const uint4*)(br[j] + kb + 16 + 8 * g); } // elems 8..15 : K = 16+8g..
  };
  auto mma = [&](FragA (&a)[2], FragA (&b)[4]) {
#pragma unroll
    for (int i = 0; i < 2; ++i)
#pragma unroll
      for (int j = 0; j < 4; ++j) acc[i][j] = wmma_bf16(a[i].v, b[j].v, acc[i][j]);
  };

  FragA a0[2], b0[4], a1[2], b1[4];
  loadAB(a0, b0, 0);
#pragma unroll 1
  for (int kb = 0; kb < K; kb += 64) {    // K % 64 == 0 for all GEMMs here
    loadAB(a1, b1, kb + 32);
    if (kb + 64 < K) {                     // gfx1250 global_prefetch_b8
      __builtin_prefetch(ar[0] + kb + 64, 0, 0);
      __builtin_prefetch(br[0] + kb + 64, 0, 0);
    }
    mma(a0, b0);
    if (kb + 64 < K) loadAB(a0, b0, kb + 64);
    mma(a1, b1);
  }

#pragma unroll
  for (int i = 0; i < 2; ++i)
#pragma unroll
    for (int j = 0; j < 4; ++j)
#pragma unroll
      for (int r = 0; r < 8; ++r) {
        size_t m = mbase + i * 16 + 8 * g + r;  // C layout: M = 8*(lane/16)+vgpr
        size_t n = nbase + j * 16 + m16;        //           N = lane%16
        float v = acc[i][j][r] + bias[n];
        if constexpr (EPI == 1) v = 0.5f * v * (1.f + erff(v * 0.7071067811865476f));
        if constexpr (EPI == 2) v += res[m * (size_t)Nn + n];
        if constexpr (EPI == 3) v *= 0.125f;   // fold 1/sqrt(dh) into Q (exact)
        C[m * (size_t)Nn + n] = (OT)v;
      }
}

// ---------------------------------------------------------------------------
// Attention: one workgroup per (b,h). Q/K/V bf16 in (B*T, D) layout, head slice
// at column h*64. Q is pre-scaled by 1/sqrt(dh) in its projection GEMM.
// Tq=Tk=150 padded to 160; padded key columns masked to -1e30, padded query
// rows clamped on load and skipped on store. src_mask adds a per-query-row
// constant before softmax => softmax-invariant => no-op.
// ---------------------------------------------------------------------------
__global__ __launch_bounds__(256) void attn_k(const bf16* __restrict__ Q,
                                              const bf16* __restrict__ Kt,
                                              const bf16* __restrict__ V,
                                              float* __restrict__ Y) {
  constexpr int T = 150, Dm = 512, DHd = 64, TP = 160;
  __shared__ __align__(32) bf16 Vt[DHd * TP];      // V^T : [d][tok]
  __shared__ __align__(32) bf16 Pl[8 * 16 * TP];   // per-wave P : [wave][m][tok]
  const int tid  = threadIdx.x;
  const int lane = tid & 31, wave = tid >> 5;
  const int m16  = lane & 15, g = lane >> 4;
  const int b = blockIdx.x >> 3;   // H = 8
  const int h = blockIdx.x & 7;

  // Stage V^T (zero the padded tokens so they can never contribute)
  for (int idx = tid; idx < DHd * TP; idx += 256) {
    int d = idx & 63, tok = idx >> 6;
    float v = 0.f;
    if (tok < T) v = (float)V[((size_t)b * T + tok) * Dm + h * DHd + d];
    Vt[d * TP + tok] = (bf16)v;
  }
  __syncthreads();

  v8f zero = {0.f, 0.f, 0.f, 0.f, 0.f, 0.f, 0.f, 0.f};

#pragma unroll 1            // keep exactly one M-tile's state live (<256 VGPRs)
  for (int mt = wave; mt < 10; mt += 8) {
    int qtok = mt * 16 + m16; if (qtok > T - 1) qtok = T - 1;
    const bf16* qp = Q + ((size_t)b * T + qtok) * Dm + h * DHd;
    FragA qa[2];
#pragma unroll
    for (int ks = 0; ks < 2; ++ks) {
      qa[ks].q[0] = *(const uint4*)(qp + 32 * ks + 8 * g);
      qa[ks].q[1] = *(const uint4*)(qp + 32 * ks + 16 + 8 * g);
    }
    // S = (Q/sqrt(dh)) K^T
    v8f s[10];
#pragma unroll
    for (int nt = 0; nt < 10; ++nt) s[nt] = zero;
#pragma unroll
    for (int nt = 0; nt < 10; ++nt) {
      int ktok = nt * 16 + m16; if (ktok > T - 1) ktok = T - 1;
      const bf16* kp = Kt + ((size_t)b * T + ktok) * Dm + h * DHd;
#pragma unroll
      for (int ks = 0; ks < 2; ++ks) {
        FragA kb;
        kb.q[0] = *(const uint4*)(kp + 32 * ks + 8 * g);
        kb.q[1] = *(const uint4*)(kp + 32 * ks + 16 + 8 * g);
        s[nt] = wmma_bf16(qa[ks].v, kb.v, s[nt]);
      }
    }
    if (144 + m16 >= T) {                               // mask padded key columns
#pragma unroll
      for (int r = 0; r < 8; ++r) s[9][r] = -1e30f;
    }
    // Row softmax (row r lives in lanes with same lane/16; reduce over lane%16)
#pragma unroll
    for (int r = 0; r < 8; ++r) {
      float mx = -3.4e38f;
#pragma unroll
      for (int nt = 0; nt < 10; ++nt) mx = fmaxf(mx, s[nt][r]);
      for (int mm = 1; mm < 16; mm <<= 1) mx = fmaxf(mx, __shfl_xor(mx, mm, 32));
      float sum = 0.f;
#pragma unroll
      for (int nt = 0; nt < 10; ++nt) {
        float e = __expf(s[nt][r] - mx);
        s[nt][r] = e; sum += e;
      }
      for (int mm = 1; mm < 16; mm <<= 1) sum += __shfl_xor(sum, mm, 32);
      float inv = 1.f / sum;
#pragma unroll
      for (int nt = 0; nt < 10; ++nt) s[nt][r] *= inv;
    }
    // Round-trip P through per-wave LDS to re-shape C-layout -> A-fragment layout
    bf16* pw = Pl + wave * 16 * TP;
#pragma unroll
    for (int r = 0; r < 8; ++r) {
      int mm = 8 * g + r;
#pragma unroll
      for (int nt = 0; nt < 10; ++nt) pw[mm * TP + nt * 16 + m16] = (bf16)s[nt][r];
    }
    // O = P @ V  (K = 160 tokens in 5 steps; N = 64 head dims in 4 tiles)
    v8f o[4];
#pragma unroll
    for (int j = 0; j < 4; ++j) o[j] = zero;
#pragma unroll
    for (int ks = 0; ks < 5; ++ks) {
      const bf16* pr = pw + m16 * TP + 32 * ks;
      FragA pa;
      pa.q[0] = *(const uint4*)(pr + 8 * g);
      pa.q[1] = *(const uint4*)(pr + 16 + 8 * g);
#pragma unroll
      for (int j = 0; j < 4; ++j) {
        FragA vb;
        vb.q[0] = *(const uint4*)(Vt + (j * 16 + m16) * TP + 32 * ks + 8 * g);
        vb.q[1] = *(const uint4*)(Vt + (j * 16 + m16) * TP + 32 * ks + 16 + 8 * g);
        o[j] = wmma_bf16(pa.v, vb.v, o[j]);
      }
    }
#pragma unroll
    for (int j = 0; j < 4; ++j)
#pragma unroll
      for (int r = 0; r < 8; ++r) {
        int tok = mt * 16 + 8 * g + r;
        if (tok < T)
          Y[((size_t)b * T + tok) * Dm + h * DHd + j * 16 + m16] = o[j][r];
      }
  }
}

// ---------------------------------------------------------------------------
// Host orchestration
// ---------------------------------------------------------------------------
static inline int ceil_div_i(int a, int b) { return (a + b - 1) / b; }

extern "C" void kernel_launch(void* const* d_in, const int* in_sizes, int n_in,
                              void* d_out, int out_size, void* d_ws, size_t ws_size,
                              hipStream_t stream) {
  (void)in_sizes; (void)n_in; (void)out_size; (void)ws_size;
  constexpr int B = 64, T = 150, D = 512, N = 150, L = 256, TE = 2048,
                FFN = 1024, LAYERS = 8, PPL = 40;
  const int BT = B * T;        // 9600 rows
  const float* x_in = (const float*)d_in[0];
  const float* xf   = (const float*)d_in[1];
  const float* emb  = (const float*)d_in[2];
  // d_in[3] = src_mask: per-query-row constant before softmax -> no-op, unused.
  auto P = [&](int l, int j) { return (const float*)d_in[4 + l * PPL + j]; };

  float* X = (float*)d_out;  // evolving residual stream lives in d_out

  // Workspace carve-up
  char* w = (char*)d_ws;
  size_t off = 0;
  auto alloc = [&](size_t bytes) -> void* {
    void* p = w + off;
    off = (off + bytes + 255) & ~(size_t)255;
    return p;
  };
  bf16*  XN   = (bf16*)alloc((size_t)BT * D * 2);
  bf16*  XB   = (bf16*)alloc((size_t)BT * D * 2);
  bf16*  XFN  = (bf16*)alloc((size_t)B * N * L * 2);
  bf16*  Qb   = (bf16*)alloc((size_t)BT * D * 2);
  bf16*  Kb   = (bf16*)alloc((size_t)BT * D * 2);
  bf16*  Vb   = (bf16*)alloc((size_t)BT * D * 2);
  float* Y    = (float*)alloc((size_t)BT * D * 4);
  bf16*  H2   = (bf16*)alloc((size_t)BT * D * 2);
  bf16*  FH   = (bf16*)alloc((size_t)BT * FFN * 2);
  float* E    = (float*)alloc((size_t)B * 2 * D * 4);
  bf16*  SEMB = (bf16*)alloc((size_t)B * TE * 2);
  bf16*  WT   = (bf16*)alloc((size_t)TE * (2 * D) * 2);  // largest weight (2048x1024)

  // x -> d_out ; silu(emb) -> bf16 once (shared by all 24 adaLN blocks)
  hipMemcpyAsync(X, x_in, (size_t)BT * D * 4, hipMemcpyDeviceToDevice, stream);
  {
    int n = B * TE;
    silu_cvt_k<<<dim3(ceil_div_i(n, 256)), dim3(256), 0, stream>>>(emb, SEMB, n);
  }

  auto LN = [&](const float* in, const float* g, const float* b_, bf16* out,
                int rows, int width) {
    ln_k<<<dim3(ceil_div_i(rows, 8)), dim3(256), 0, stream>>>(in, g, b_, out, rows, width);
  };
  auto WTR = [&](const float* Wp, int Kd, int Nd) {
    int n = Kd * Nd;
    wtrans_k<<<dim3(ceil_div_i(n, 256)), dim3(256), 0, stream>>>(Wp, WT, Kd, Nd);
  };
  auto GEMM_B = [&](const bf16* A, const float* bias, bf16* C, int M, int Kd, int Nd) {
    gemm_k<0, bf16><<<dim3(Nd / 256, M / 64), dim3(256), 0, stream>>>(
        A, WT, bias, C, nullptr, M, Kd, Nd);
  };
  auto GEMM_Q = [&](const bf16* A, const float* bias, bf16* C, int M, int Kd, int Nd) {
    gemm_k<3, bf16><<<dim3(Nd / 256, M / 64), dim3(256), 0, stream>>>(
        A, WT, bias, C, nullptr, M, Kd, Nd);
  };
  auto GEMM_F = [&](const bf16* A, const float* bias, float* C, int M, int Kd, int Nd) {
    gemm_k<0, float><<<dim3(Nd / 256, M / 64), dim3(256), 0, stream>>>(
        A, WT, bias, C, nullptr, M, Kd, Nd);
  };
  auto GEMM_G = [&](const bf16* A, const float* bias, bf16* C, int M, int Kd, int Nd) {
    gemm_k<1, bf16><<<dim3(Nd / 256, M / 64), dim3(256), 0, stream>>>(
        A, WT, bias, C, nullptr, M, Kd, Nd);
  };
  auto GEMM_R = [&](const bf16* A, const float* bias, float* C, int M, int Kd, int Nd) {
    gemm_k<2, float><<<dim3(Nd / 256, M / 64), dim3(256), 0, stream>>>(
        A, WT, bias, C, C, M, Kd, Nd);
  };
  auto ADALN = [&](const float* embW, const float* embB, const float* lng,
                   const float* lnb, const float* outW, const float* outB) {
    WTR(embW, TE, 2 * D);
    GEMM_F(SEMB, embB, E, B, TE, 2 * D);     // (64,2048)@(2048,1024)
    modulate_k<<<dim3(ceil_div_i(BT, 8)), dim3(256), 0, stream>>>(Y, E, lng, lnb, H2, BT, T);
    WTR(outW, D, D);
    GEMM_R(H2, outB, X, BT, D, D);           // x += silu(...) @ out_W + out_b
  };

  for (int l = 0; l < LAYERS; ++l) {
    // ---- temporal self-attention ----
    LN(X, P(l, 0), P(l, 1), XN, BT, D);
    WTR(P(l, 2), D, D);  GEMM_Q(XN, P(l, 3), Qb, BT, D, D);   // Q pre-scaled by 1/8
    WTR(P(l, 4), D, D);  GEMM_B(XN, P(l, 5), Kb, BT, D, D);
    WTR(P(l, 6), D, D);  GEMM_B(XN, P(l, 7), Vb, BT, D, D);
    attn_k<<<dim3(B * 8), dim3(256), 0, stream>>>(Qb, Kb, Vb, Y);
    ADALN(P(l, 8), P(l, 9), P(l, 12), P(l, 13), P(l, 10), P(l, 11));
    // ---- temporal cross-attention ----
    LN(X, P(l, 14), P(l, 15), XN, BT, D);
    LN(xf, P(l, 16), P(l, 17), XFN, B * N, L);
    WTR(P(l, 18), D, D); GEMM_Q(XN,  P(l, 19), Qb, BT, D, D); // Q pre-scaled by 1/8
    WTR(P(l, 20), L, D); GEMM_B(XFN, P(l, 21), Kb, B * N, L, D);
    WTR(P(l, 22), L, D); GEMM_B(XFN, P(l, 23), Vb, B * N, L, D);
    attn_k<<<dim3(B * 8), dim3(256), 0, stream>>>(Qb, Kb, Vb, Y);
    ADALN(P(l, 24), P(l, 25), P(l, 28), P(l, 29), P(l, 26), P(l, 27));
    // ---- FFN (note: uses raw x, not LN(x), per reference) ----
    {
      int n = BT * D;
      cvt_k<<<dim3(ceil_div_i(n, 256)), dim3(256), 0, stream>>>(X, XB, n);
    }
    WTR(P(l, 30), D, FFN); GEMM_G(XB, P(l, 31), FH, BT, D, FFN);
    WTR(P(l, 32), FFN, D); GEMM_F(FH, P(l, 33), Y, BT, FFN, D);
    ADALN(P(l, 34), P(l, 35), P(l, 38), P(l, 39), P(l, 36), P(l, 37));
  }
  // final x already resides in d_out
}
